// TemporalGNN_40209483825425
// MI455X (gfx1250) — matrix operand order repaired and checked by reading, not compiled, not run on hardware
//
#include <hip/hip_runtime.h>
#include <math.h>
#include <stdint.h>

// ---------------------------------------------------------------------------
// TemporalGNN (TGCN-style GRU over GCN with C=1) for MI455X / gfx1250.
// Memory-bound: x = 491MB read once; y (46MB) lives in the 192MB L2.
// Projection: v_wmma_f32_16x16x32_f16 (K=32 == F in one instruction),
// x tile staged via global_load_async_to_lds_b128 (ASYNCcnt, zero VGPR cost).
// ---------------------------------------------------------------------------

typedef __attribute__((ext_vector_type(16))) _Float16 v16h;
typedef __attribute__((ext_vector_type(8)))  float    v8f;

#define TT 12            // periods
#define FF 32            // features
#define BB 32            // batch == wave32 lanes
#define FT (FF * TT)     // 384 contiguous floats per (b,n) row of x
#define FTP (FT + 4)     // padded LDS row stride (16B-aligned, bank-skewed)

// --------------------------- graph preprocessing ---------------------------

__global__ void zero_int_kernel(int* __restrict__ p, int n) {
    int i = blockIdx.x * blockDim.x + threadIdx.x;
    if (i < n) p[i] = 0;
}

__global__ void count_kernel(const int* __restrict__ dst, int* __restrict__ count, int E) {
    int e = blockIdx.x * blockDim.x + threadIdx.x;
    if (e < E) atomicAdd(&count[dst[e]], 1);
}

// single-block scan: CSR offsets (exclusive) + dinv = rsqrt(deg), deg = count+1 (self loop)
__global__ void scan_kernel(const int* __restrict__ count, int* __restrict__ offsets,
                            float* __restrict__ dinv, int N) {
    __shared__ int buf[1024];
    __shared__ int carry;
    if (threadIdx.x == 0) carry = 0;
    __syncthreads();
    for (int base = 0; base < N; base += 1024) {
        int i = base + (int)threadIdx.x;
        int v = (i < N) ? count[i] : 0;
        buf[threadIdx.x] = v;
        __syncthreads();
        for (int off = 1; off < 1024; off <<= 1) {
            int t = ((int)threadIdx.x >= off) ? buf[threadIdx.x - off] : 0;
            __syncthreads();
            buf[threadIdx.x] += t;
            __syncthreads();
        }
        int incl = buf[threadIdx.x];
        int c = carry;
        if (i < N) {
            offsets[i] = c + incl - v;               // exclusive prefix
            dinv[i] = rsqrtf((float)(v + 1));        // deg >= 1 always (self loop)
        }
        __syncthreads();
        if (threadIdx.x == 1023) carry = c + buf[1023];
        __syncthreads();
    }
    if (threadIdx.x == 0) offsets[N] = carry;
}

__global__ void fill_kernel(const int* __restrict__ src, const int* __restrict__ dst,
                            const int* __restrict__ offsets, int* __restrict__ cursor,
                            const float* __restrict__ dinv,
                            int* __restrict__ csr_src, float* __restrict__ csr_w, int E) {
    int e = blockIdx.x * blockDim.x + threadIdx.x;
    if (e < E) {
        int sn = src[e], dn = dst[e];
        int pos = offsets[dn] + atomicAdd(&cursor[dn], 1);
        csr_src[pos] = sn;
        csr_w[pos] = dinv[sn] * dinv[dn];
    }
}

// --------------------------- WMMA projection -------------------------------
// y[g][t][n][b] = dot(x[b,n,:,t], W_g),  g in {z,r,h}
// One wave handles (node n, 16 batches b0..b0+15), 12 timesteps = 12 WMMAs.
// A (16x32 f16) = gate weights in rows 0..2 (rest zero), reused all t.
// B (32x16 f16) = x^T tile for timestep t, built from an LDS-staged x tile.
// x tile (16 rows x 1536B) staged with global_load_async_to_lds_b128: the
// async engine writes LDS directly, so no load-data VGPRs are allocated.

__global__ __launch_bounds__(64) void proj_kernel(
        const float* __restrict__ x,
        const float* __restrict__ wz, const float* __restrict__ wr,
        const float* __restrict__ wh,
        float* __restrict__ y, int N) {
    __shared__ __align__(16) float lds[2][16 * FTP];
    const int lane = threadIdx.x & 31;
    const int wave = threadIdx.x >> 5;
    const int waveGlobal = blockIdx.x * 2 + wave;
    const int n  = waveGlobal >> 1;
    const int b0 = (waveGlobal & 1) << 4;    // 0 or 16
    if (n >= N) return;                       // wave-uniform (EXEC stays full)
    float* sm = &lds[wave][0];

    // stage 16 rows x 384 contiguous floats of x: async global -> LDS, 16B/lane
    for (int idx4 = lane; idx4 < 16 * (FT / 4); idx4 += 32) {
        int row = idx4 / (FT / 4);
        int c4  = idx4 % (FT / 4);
        const float* gptr = x + ((size_t)(b0 + row) * N + n) * FT + (size_t)c4 * 4;
        uint32_t laddr = (uint32_t)(uintptr_t)(sm + row * FTP + c4 * 4);
        asm volatile("global_load_async_to_lds_b128 %0, %1, off"
                     :: "v"(laddr), "v"(gptr) : "memory");
    }
    asm volatile("s_wait_asynccnt 0" ::: "memory");
    __syncthreads();

    // A fragment: A[M][K] = W_M[K] for M<3 else 0.  contiguous k order (same as B):
    // lane = m + 16h ; element i -> K = 16h + i
    v16h a;
    {
        const int m = lane & 15, h = lane >> 4;
        const float* W = (m == 0) ? wz : (m == 1) ? wr : (m == 2) ? wh : nullptr;
        #pragma unroll
        for (int i = 0; i < 16; ++i) {
            int K = 16 * h + i;
            float v = (m < 3) ? W[K] : 0.0f;
            a[i] = (_Float16)v;
        }
    }

    const int col   = lane & 15;            // B column = batch offset
    const int kbase = (lane >> 4) << 4;     // contiguous k order (as A): lanes 0-15 K=0..15, 16-31 K=16..31
    const size_t plane = (size_t)TT * N * BB;   // stride between gate planes of y

    for (int t = 0; t < TT; ++t) {
        v16h bfrag;
        #pragma unroll
        for (int i = 0; i < 16; ++i) {
            float v = sm[col * FTP + (kbase + i) * TT + t];
            bfrag[i] = (_Float16)v;
        }
        v8f c = {};
        c = __builtin_amdgcn_wmma_f32_16x16x32_f16(
                /*neg_a=*/false, a, /*neg_b=*/false, bfrag,
                /*c_mod=*/(short)0, c, /*reuse_a=*/false, /*reuse_b=*/false);
        if (lane < 16) {
            // D: lanes 0-15 hold N=lane, VGPR j = row M=j  ->  c[g] = gate g
            size_t base = ((size_t)t * N + n) * BB + b0 + lane;   // g = 0 plane
            y[base]             = c[0];
            y[base + plane]     = c[1];
            y[base + 2 * plane] = c[2];
        }
    }
}

// ------------------- fused SpMM + GRU + attention + output -----------------
// Wave per node n, lane per batch b (B == 32 == wave32): every y access is a
// contiguous 128B line; y (46MB) lives in the 192MB L2 across 33x reuse.

__global__ __launch_bounds__(256) void gru_kernel(
        const float* __restrict__ y,
        const int* __restrict__ offsets, const int* __restrict__ csr_src,
        const float* __restrict__ csr_w, const float* __restrict__ dinv,
        const float* __restrict__ att,
        const float* __restrict__ czb, const float* __restrict__ crb,
        const float* __restrict__ chb,
        const float* __restrict__ lzw, const float* __restrict__ lzb,
        const float* __restrict__ lrw, const float* __restrict__ lrb,
        const float* __restrict__ lhw, const float* __restrict__ lhb,
        float* __restrict__ out, int N) {
    const int n = (int)((blockIdx.x * blockDim.x + threadIdx.x) >> 5);
    const int b = threadIdx.x & 31;
    if (n >= N) return;

    // softmax over the 12 attention logits (tiny, recomputed per thread)
    float p[TT], mx = -1e30f;
    #pragma unroll
    for (int t = 0; t < TT; ++t) { p[t] = att[t]; mx = fmaxf(mx, p[t]); }
    float s = 0.0f;
    #pragma unroll
    for (int t = 0; t < TT; ++t) { p[t] = __expf(p[t] - mx); s += p[t]; }
    const float pinv = 1.0f / s;

    const float cz = czb[0], cr = crb[0], ch = chb[0];
    const float lzw0 = lzw[0], lzw1 = lzw[1], lzb0 = lzb[0];
    const float lrw0 = lrw[0], lrw1 = lrw[1], lrb0 = lrb[0];
    const float lhw0 = lhw[0], lhw1 = lhw[1], lhb0 = lhb[0];

    const int e0 = offsets[n], e1 = offsets[n + 1];
    const float selfw = dinv[n] * dinv[n];          // self-loop norm
    const size_t plane = (size_t)TT * N * BB;

    float H = 0.0f, acc = 0.0f;
    for (int t = 0; t < TT; ++t) {
        const float* yz = y + (size_t)t * N * BB;
        const float* yr = yz + plane;
        const float* yh = yr + plane;
        const size_t selfoff = (size_t)n * BB + b;
        float az = selfw * yz[selfoff];
        float ar = selfw * yr[selfoff];
        float ah = selfw * yh[selfoff];
        for (int e = e0; e < e1; ++e) {
            int   sn = csr_src[e];                  // uniform across wave
            float w  = csr_w[e];
            size_t off = (size_t)sn * BB + b;       // coalesced 128B per plane
            az += w * yz[off];
            ar += w * yr[off];
            ah += w * yh[off];
        }
        az += cz; ar += cr; ah += ch;
        float Z  = 1.0f / (1.0f + __expf(-(lzw0 * az + lzw1 * H + lzb0)));
        float R  = 1.0f / (1.0f + __expf(-(lrw0 * ar + lrw1 * H + lrb0)));
        float Ht = tanhf(lhw0 * ah + lhw1 * (H * R) + lhb0);
        H = Z * H + (1.0f - Z) * Ht;
        acc += (p[t] * pinv) * H;
    }
    float hr = fmaxf(acc, 0.0f);
    out[(size_t)b * N + n] = 1.0f / (1.0f + __expf(-hr));
}

// ------------------------------- launcher ----------------------------------

extern "C" void kernel_launch(void* const* d_in, const int* in_sizes, int n_in,
                              void* d_out, int out_size, void* d_ws, size_t ws_size,
                              hipStream_t stream) {
    (void)n_in; (void)ws_size;
    const float* x        = (const float*)d_in[0];
    const int*   edge     = (const int*)  d_in[1];
    const float* conv_z_w = (const float*)d_in[2];
    const float* conv_z_b = (const float*)d_in[3];
    const float* conv_r_w = (const float*)d_in[4];
    const float* conv_r_b = (const float*)d_in[5];
    const float* conv_h_w = (const float*)d_in[6];
    const float* conv_h_b = (const float*)d_in[7];
    const float* lin_z_w  = (const float*)d_in[8];
    const float* lin_z_b  = (const float*)d_in[9];
    const float* lin_r_w  = (const float*)d_in[10];
    const float* lin_r_b  = (const float*)d_in[11];
    const float* lin_h_w  = (const float*)d_in[12];
    const float* lin_h_b  = (const float*)d_in[13];
    const float* att      = (const float*)d_in[14];

    const int N = out_size / BB;          // 10000
    const int E = in_sizes[1] / 2;        // 320000
    const int* srcA = edge;
    const int* dstA = edge + E;

    // workspace carve (256B aligned)
    char* p = (char*)d_ws;
    auto carve = [&](size_t bytes) {
        char* r = p;
        p += (bytes + 255) & ~(size_t)255;
        return r;
    };
    float* y       = (float*)carve(sizeof(float) * 3 * TT * (size_t)N * BB); // 46MB
    float* dinv    = (float*)carve(sizeof(float) * N);
    int*   count   = (int*)  carve(sizeof(int) * 2 * (size_t)N);   // count + cursor
    int*   cursor  = count + N;
    int*   offsets = (int*)  carve(sizeof(int) * (N + 1));
    int*   csr_src = (int*)  carve(sizeof(int) * (size_t)E);
    float* csr_w   = (float*)carve(sizeof(float) * (size_t)E);

    zero_int_kernel<<<(2 * N + 255) / 256, 256, 0, stream>>>(count, 2 * N);
    count_kernel<<<(E + 255) / 256, 256, 0, stream>>>(dstA, count, E);
    scan_kernel<<<1, 1024, 0, stream>>>(count, offsets, dinv, N);
    fill_kernel<<<(E + 255) / 256, 256, 0, stream>>>(srcA, dstA, offsets, cursor,
                                                     dinv, csr_src, csr_w, E);
    proj_kernel<<<N, 64, 0, stream>>>(x, conv_z_w, conv_r_w, conv_h_w, y, N);
    gru_kernel<<<(N + 7) / 8, 256, 0, stream>>>(y, offsets, csr_src, csr_w, dinv, att,
                                                conv_z_b, conv_r_b, conv_h_b,
                                                lin_z_w, lin_z_b, lin_r_w, lin_r_b,
                                                lin_h_w, lin_h_b,
                                                (float*)d_out, N);
}
